// RNNBaseline_41669772705987
// MI455X (gfx1250) — compile-verified, matrix operand order checked by reading, not run on hardware
//
#include <hip/hip_runtime.h>
#include <hip/hip_bf16.h>
#include <stdint.h>

// ---------------------------------------------------------------------------
// GRU forward for MI455X (gfx1250, wave32, WMMA).
//   gi = x@Wi + bi            (bf16 WMMA GEMM, f32 acc, software-pipelined)
//   scan over S: gh = h@Wh    (16 persistent WGs, Wh slice in LDS via TDM,
//                              per-step h broadcast via async-to-LDS loads)
//   out = rnn@Wo + bo         (bf16 WMMA GEMM, f32 acc)
// All matrix operands are pre-swizzled into the exact ISA 16x16x32 bf16
// fragment layout so every fragment load is 32 contiguous bytes per lane.
// ---------------------------------------------------------------------------

typedef unsigned short u16;
typedef __attribute__((ext_vector_type(16))) __bf16        v16bf;
typedef __attribute__((ext_vector_type(8)))  float         v8f;
typedef __attribute__((ext_vector_type(8)))  unsigned int  v8u;
typedef __attribute__((ext_vector_type(4)))  unsigned int  v4u;

constexpr int BB = 16, SS = 4096, DD = 512;
constexpr int MR = BB * SS;      // 65536 rows of x / rnn
constexpr int N3 = 3 * DD;       // 1536
constexpr int GQ = 16;           // scan groups (one WGP each)
constexpr int JG = DD / GQ;      // 32 hidden units per group
constexpr int NC = 3 * JG;       // 96 packed gh columns per group

// ---- workspace layout (bytes); requires ~516 MB of d_ws -------------------
constexpr size_t XE_OFF  = 0;                         // x bf16, A-frag order
constexpr size_t XE_SZ   = (size_t)MR * DD * 2;       // 64 MB
constexpr size_t WIE_OFF = XE_OFF + XE_SZ;            // Wi bf16, B-frag order
constexpr size_t WIE_SZ  = (size_t)DD * N3 * 2;
constexpr size_t WOE_OFF = WIE_OFF + WIE_SZ;          // Wo bf16, B-frag order
constexpr size_t WOE_SZ  = (size_t)DD * DD * 2;
constexpr size_t WHP_OFF = WOE_OFF + WOE_SZ;          // Wh bf16, packed/frag
constexpr size_t WHP_SZ  = (size_t)DD * N3 * 2;
constexpr size_t GI_OFF  = WHP_OFF + WHP_SZ;          // gi f32 [MR][1536]
constexpr size_t GI_SZ   = (size_t)MR * N3 * 4;       // 384 MB
constexpr size_t RNN_OFF = GI_OFF + GI_SZ;            // rnn bf16, A-frag order
constexpr size_t RNN_SZ  = (size_t)MR * DD * 2;       // 64 MB
constexpr size_t HB_OFF  = RNN_OFF + RNN_SZ;          // h double buffer (frag)
constexpr size_t HB_SZ   = (size_t)2 * BB * DD * 2;   // 32 KB
constexpr size_t CNT_OFF = HB_OFF + HB_SZ;            // (SS+1) step counters

// ---- helpers ---------------------------------------------------------------
__device__ __forceinline__ u16 f2bf(float f) {          // RNE f32 -> bf16
  unsigned u = __builtin_bit_cast(unsigned, f);
  u += 0x7FFFu + ((u >> 16) & 1u);
  return (u16)(u >> 16);
}
__device__ __forceinline__ float bf2f(u16 h) {
  unsigned u = ((unsigned)h) << 16;
  return __builtin_bit_cast(float, u);
}
// ISA 16-bit A/B fragment k index for vector element e (0..15), lane l.
__device__ __forceinline__ int kmap(int e, int l) {
  return e + ((e >> 3) << 3) + ((l >> 4) << 3);
}
__device__ __forceinline__ v16bf ldfrag(const u16* p) {
  v8u r = *(const v8u*)p;                // 32 B / lane -> 2x b128
  return __builtin_bit_cast(v16bf, r);
}
__device__ __forceinline__ v16bf zfrag() {
  return __builtin_bit_cast(v16bf, (v8u)0);
}
__device__ __forceinline__ v8f wmma_bf16(v16bf a, v16bf b, v8f c) {
  return __builtin_amdgcn_wmma_f32_16x16x32_bf16(false, a, false, b,
                                                 (short)0, c, false, false);
}
// element (row m, k=d) inside one 16x512 A-frag tile (flat half index)
__device__ __forceinline__ int hfragidx(int d, int m) {
  int dk = d & 31;
  int lh = (dk >> 3) & 1;
  int e  = (dk & 7) + ((dk & 16) ? 8 : 0);
  return (((d >> 5) * 32) + lh * 16 + m) * 16 + e;
}

// ---- prep kernels ----------------------------------------------------------
__global__ void k_init(unsigned* hbuf32, unsigned* cnt) {
  int i = blockIdx.x * blockDim.x + threadIdx.x;
  if (i < 8192) hbuf32[i] = 0u;                       // both h buffers = 0
  int j = i - 8192;
  if (j >= 0 && j <= SS) cnt[j] = (j == 0) ? (unsigned)GQ : 0u;
}

// x f32 [MR][512] -> A-frag order bf16 [MR/16][16kk][32lane][16e]
__global__ void k_pack_a(const float* __restrict__ x, u16* __restrict__ dst) {
  size_t u = (size_t)blockIdx.x * blockDim.x + threadIdx.x;
  int e = u & 15, l = (u >> 4) & 31, kk = (u >> 9) & 15;
  size_t mt = u >> 13;
  size_t row = mt * 16 + (l & 15);
  int k = kk * 32 + kmap(e, l);
  dst[u] = f2bf(x[row * DD + k]);
}

// W f32 [512][N] -> B-frag order bf16 [N/16][16kk][32lane][16e]
__global__ void k_pack_b(const float* __restrict__ W, u16* __restrict__ dst, int N) {
  size_t u = (size_t)blockIdx.x * blockDim.x + threadIdx.x;
  int e = u & 15, l = (u >> 4) & 31, kk = (u >> 9) & 15;
  int nt = (int)(u >> 13);
  int n = nt * 16 + (l & 15);
  int k = kk * 32 + kmap(e, l);
  dst[u] = f2bf(W[(size_t)k * N + n]);
}

// Wh f32 [512][1536] -> per-group packed r|z|n B-frag order:
// [g][tile=kk*6+nn][32lane][16e], group g owns hidden units g*32..g*32+31.
__global__ void k_pack_wh(const float* __restrict__ Wh, u16* __restrict__ dst) {
  size_t u = (size_t)blockIdx.x * blockDim.x + threadIdx.x;
  int e = u & 15, l = (u >> 4) & 31;
  int rem = (int)(u >> 9);
  int tile = rem % 96, g = rem / 96;
  int kk = tile / 6, nn = tile % 6;
  int k = kk * 32 + kmap(e, l);
  int c96 = nn * 16 + (l & 15);
  int p = c96 >> 5, j = c96 & 31;
  int col = p * DD + g * JG + j;
  dst[u] = f2bf(Wh[(size_t)k * N3 + col]);
}

// ---- generic bf16 WMMA GEMM: C[Mtiles*16][N] = Afrag @ Bfrag + bias --------
// one wave computes a 16x64 tile (4 accumulators share the A fragment);
// 1-deep software pipeline so loads for kk+1 overlap the WMMAs of kk.
__global__ __launch_bounds__(256) void k_gemm(const u16* __restrict__ Af,
                                              const u16* __restrict__ Bf,
                                              const float* __restrict__ bias,
                                              float* __restrict__ C,
                                              int Mtiles, int N) {
  int lane = threadIdx.x & 31;
  int gw = blockIdx.x * (blockDim.x >> 5) + (threadIdx.x >> 5);
  int nt64 = N >> 6;
  int mt = gw / nt64, nt = gw % nt64;
  if (mt >= Mtiles) return;

  const u16* Ab = Af + ((size_t)mt * 16) * 512 + lane * 16;
  const u16* Bb = Bf + ((size_t)nt * 64) * 512 + lane * 16;
  v8f acc[4] = {v8f{}, v8f{}, v8f{}, v8f{}};

  v16bf aC = ldfrag(Ab);
  v16bf bC[4];
#pragma unroll
  for (int t4 = 0; t4 < 4; t4++) bC[t4] = ldfrag(Bb + (size_t)(t4 * 16) * 512);

#pragma unroll
  for (int kk = 0; kk < 16; kk++) {
    v16bf aN = zfrag();
    v16bf bN[4] = {zfrag(), zfrag(), zfrag(), zfrag()};
    if (kk < 15) {
      aN = ldfrag(Ab + (size_t)(kk + 1) * 512);
#pragma unroll
      for (int t4 = 0; t4 < 4; t4++)
        bN[t4] = ldfrag(Bb + (size_t)(t4 * 16 + kk + 1) * 512);
      __builtin_prefetch(Ab + (size_t)(kk + 2) * 512, 0, 3);
    }
#pragma unroll
    for (int t4 = 0; t4 < 4; t4++) acc[t4] = wmma_bf16(aC, bC[t4], acc[t4]);
    aC = aN;
#pragma unroll
    for (int t4 = 0; t4 < 4; t4++) bC[t4] = bN[t4];
  }

  int r0 = mt * 16 + ((lane >> 4) << 3);  // m = v + 8*(lane/16)
  int n0 = nt * 64 + (lane & 15);
#pragma unroll
  for (int t4 = 0; t4 < 4; t4++) {
    int n = n0 + t4 * 16;
    float bn = bias[n];
#pragma unroll
    for (int v = 0; v < 8; v++)
      C[(size_t)(r0 + v) * N + n] = acc[t4][v] + bn;
  }
}

// ---- persistent GRU scan ---------------------------------------------------
// 16 workgroups x 192 threads (6 waves). WG g keeps Wh slice (512x96 bf16,
// 96 KB) in LDS (loaded once by the Tensor Data Mover); each step:
// h_t broadcast via global_load_async_to_lds_b128, gh = h@Wh_slice via WMMA,
// gates, publish h slice through an L2 double buffer + counter barrier.
__global__ __launch_bounds__(192, 1) void k_scan(const u16* __restrict__ whp,
                                                 const float* __restrict__ gi,
                                                 const float* __restrict__ bhn,
                                                 u16* __restrict__ hbuf,
                                                 u16* __restrict__ rnn,
                                                 unsigned* __restrict__ cnt) {
  extern __shared__ __align__(16) char smem[];
  u16*   WhS = (u16*)smem;                               // 96*512 halves
  u16*   hS  = (u16*)(smem + (size_t)NC * DD * 2);       // 16*512 halves (frag)
  float* ghS = (float*)(smem + (size_t)NC * DD * 2 + (size_t)BB * DD * 2);

  const int g = blockIdx.x, tid = threadIdx.x;
  const int lane = tid & 31, wv = tid >> 5;  // wv = n-tile 0..5

  // ---- one-shot Wh slice load via the Tensor Data Mover (wave 0 only) ----
  if (wv == 0) {
    unsigned whs_lds = (unsigned)(size_t)WhS;  // LDS byte offset (addr[31:0])
    unsigned long long ga =
        (unsigned long long)(size_t)(whp + (size_t)g * NC * DD);
    constexpr unsigned ELEMS = (unsigned)(NC * DD * 2 / 8);  // 12288 x 8B
    // D# group0: count=1, lds_addr, global_addr[56:0], type=2
    v4u d0 = {1u, whs_lds, (unsigned)ga,
              (unsigned)((ga >> 32) & 0x01FFFFFFu) | (2u << 30)};
    // D# group1: data_size=8B, 1-row tile: tile_dim0=tensor_dim0=12288
    v8u d1 = {0x00030000u,              // wg_mask=0, data_size=3 (8B)
              (ELEMS & 0xFFFFu) << 16,  // tensor_dim0[15:0]  -> bits 63:48
              (ELEMS >> 16) | (1u << 16),  // tensor_dim0[31:16], tensor_dim1=1
              (ELEMS & 0xFFFFu) << 16,  // tile_dim0 -> bits 127:112
              0u,                       // tile_dim1=0 (unused), tile_dim2=0
              ELEMS,                    // tensor_dim0_stride[31:0]
              0u, 0u};
    asm volatile("tensor_load_to_lds %0, %1, null, null"
                 :: "s"(d0), "s"(d1) : "memory");
    __builtin_amdgcn_s_wait_tensorcnt((short)0);
  }
  __syncthreads();

  for (int t = 0; t < SS; t++) {
    if (tid == 0) {  // wait until all 16 slices of h_t are published
      while (__hip_atomic_load(&cnt[t], __ATOMIC_ACQUIRE,
                               __HIP_MEMORY_SCOPE_AGENT) < (unsigned)GQ)
        __builtin_amdgcn_s_sleep(1);
    }
    __syncthreads();

    { // h_t (fragment order, 16 KB) -> LDS via async copy, from L2 (DEV scope)
      unsigned hs_lds = (unsigned)(size_t)hS;
      const char* src = (const char*)(hbuf + (size_t)(t & 1) * BB * DD);
      for (int i = tid; i < BB * DD * 2 / 16; i += 192) {
        asm volatile(
            "global_load_async_to_lds_b128 %0, %1, off scope:SCOPE_DEV"
            :: "v"(hs_lds + i * 16),
               "v"((unsigned long long)(size_t)(src + (size_t)i * 16))
            : "memory");
      }
      asm volatile("s_wait_asynccnt 0x0" ::: "memory");
    }
    __syncthreads();

    // gh tile: 16 (batch) x 16 cols, wave wv owns packed cols wv*16..wv*16+15
    v8f acc = v8f{};
    v16bf a0 = ldfrag(hS + (0 * 32 + lane) * 16);
    v16bf b0 = ldfrag(WhS + ((0 * 6 + wv) * 32 + lane) * 16);
#pragma unroll
    for (int kk = 0; kk < 16; kk++) {
      v16bf a1 = zfrag(), b1 = zfrag();
      if (kk < 15) {
        a1 = ldfrag(hS + ((kk + 1) * 32 + lane) * 16);
        b1 = ldfrag(WhS + (((kk + 1) * 6 + wv) * 32 + lane) * 16);
      }
      acc = wmma_bf16(a0, b0, acc);
      a0 = a1; b0 = b1;
    }
    {
      int m0 = (lane >> 4) << 3, c = wv * 16 + (lane & 15);
#pragma unroll
      for (int v = 0; v < 8; v++) ghS[(m0 + v) * NC + c] = acc[v];
    }
    __syncthreads();

    // gates + h update for this group's 32 hidden units, all 16 batches
    for (int i = tid; i < BB * JG; i += 192) {
      int b = i >> 5, j = i & 31;
      int d = g * JG + j;
      size_t gib = ((size_t)b * SS + t) * N3;
      float gr = gi[gib + d], gz = gi[gib + DD + d], gn = gi[gib + 2 * DD + d];
      float hr = ghS[b * NC + j], hz = ghS[b * NC + JG + j],
            hn = ghS[b * NC + 2 * JG + j];
      float hp = bf2f(hS[hfragidx(d, b)]);
      float r = 1.f / (1.f + __expf(-(gr + hr)));
      float z = 1.f / (1.f + __expf(-(gz + hz)));
      float n = tanhf(gn + r * (hn + bhn[d]));
      float hv = (1.f - z) * n + z * hp;
      u16 hb = f2bf(hv);
      hbuf[(size_t)((t + 1) & 1) * BB * DD + hfragidx(d, b)] = hb;
      // write rnn directly in A-frag order for the output GEMM
      int mt = b * (SS / 16) + (t >> 4);
      int dk = d & 31, lh = (dk >> 3) & 1, e = (dk & 7) + ((dk & 16) ? 8 : 0);
      rnn[(((size_t)(mt * 16 + (d >> 5))) * 32 + lh * 16 + (t & 15)) * 16 + e] = hb;
    }
    __threadfence();   // device-scope release of h slice + rnn
    __syncthreads();
    if (tid == 0)
      __hip_atomic_fetch_add(&cnt[t + 1], 1u, __ATOMIC_RELEASE,
                             __HIP_MEMORY_SCOPE_AGENT);
  }
}

// ---------------------------------------------------------------------------
extern "C" void kernel_launch(void* const* d_in, const int* in_sizes, int n_in,
                              void* d_out, int out_size, void* d_ws, size_t ws_size,
                              hipStream_t stream) {
  const float* x   = (const float*)d_in[0];
  const float* Wi  = (const float*)d_in[1];
  const float* bi  = (const float*)d_in[2];
  const float* Wh  = (const float*)d_in[3];
  const float* bhn = (const float*)d_in[4];
  const float* Wo  = (const float*)d_in[5];
  const float* bo  = (const float*)d_in[6];
  float* out = (float*)d_out;

  char* ws = (char*)d_ws;
  u16*      xe   = (u16*)(ws + XE_OFF);
  u16*      wie  = (u16*)(ws + WIE_OFF);
  u16*      woe  = (u16*)(ws + WOE_OFF);
  u16*      whp  = (u16*)(ws + WHP_OFF);
  float*    gi   = (float*)(ws + GI_OFF);
  u16*      rnn  = (u16*)(ws + RNN_OFF);
  u16*      hbuf = (u16*)(ws + HB_OFF);
  unsigned* cnt  = (unsigned*)(ws + CNT_OFF);

  // allow >64KB dynamic LDS for the scan kernel
  (void)hipFuncSetAttribute((const void*)k_scan,
                            hipFuncAttributeMaxDynamicSharedMemorySize, 131072);

  k_init<<<(8192 + SS + 1 + 255) / 256, 256, 0, stream>>>((unsigned*)hbuf, cnt);
  k_pack_a<<<(int)((size_t)MR * DD / 256), 256, 0, stream>>>(x, xe);
  k_pack_b<<<(int)((size_t)DD * N3 / 256), 256, 0, stream>>>(Wi, wie, N3);
  k_pack_b<<<(int)((size_t)DD * DD / 256), 256, 0, stream>>>(Wo, woe, DD);
  k_pack_wh<<<(int)((size_t)DD * N3 / 256), 256, 0, stream>>>(Wh, whp);

  // gi = x@Wi + bi : 4096 m-tiles x 24 n64-tiles = 98304 waves / 8 per block
  k_gemm<<<(MR / 16) * (N3 / 64) / 8, 256, 0, stream>>>(xe, wie, bi, gi,
                                                        MR / 16, N3);
  // sequential GRU scan: 16 persistent WGs, 120832 B dynamic LDS each
  k_scan<<<GQ, 192, (size_t)NC * DD * 2 + (size_t)BB * DD * 2 + BB * NC * 4,
           stream>>>(whp, gi, bhn, hbuf, rnn, cnt);
  // out = rnn@Wo + bo
  k_gemm<<<(MR / 16) * (DD / 64) / 8, 256, 0, stream>>>(rnn, woe, bo, out,
                                                        MR / 16, DD);
}
